// ScannedLSTM_37684043055511
// MI455X (gfx1250) — matrix operand order repaired and checked by reading, not compiled, not run on hardware
//
#include <hip/hip_runtime.h>

// Scanned LSTM on gfx1250 (MI455X): persistent cooperative kernel.
// - f16 WMMA 16x16x32 (v_wmma_f32_16x16x32_f16), f32 accumulation, f32 cell state in registers
// - per-block weight slice resident in LDS ([kb][matrix][gate][lane][16], 128 KB) all 512 steps
// - x pre-converted to f16 one step ahead (cooperative, ping-pong), h f16 ping-pong
// - B fragments software-pipelined; one atomic grid barrier per timestep

#define TT   512
#define BB   256
#define HH   512
#define G4H  2048
#define NBLK 64
#define NTHR 256
#define KB   16      // k-blocks of 32 over K=512
#define LDS_BYTES 131072

typedef __attribute__((ext_vector_type(16))) _Float16 v16h;
typedef __attribute__((ext_vector_type(2)))  _Float16 v2h;
typedef __attribute__((ext_vector_type(8)))  float    v8f;

union V16U {
    uint4          u[2];
    unsigned       q[8];
    unsigned short s[16];
    v16h           v;
};

// f32 -> f16 (hardware cvt, RNE)
__device__ __forceinline__ unsigned short f2h(float f) {
    _Float16 h = (_Float16)f;
    return __builtin_bit_cast(unsigned short, h);
}

// pack two f32 into two f16 in one dword (v_cvt_f16_f32 + v_pack)
__device__ __forceinline__ unsigned pk2h(float lo, float hi) {
    v2h p;
    p.x = (_Float16)lo;
    p.y = (_Float16)hi;
    return __builtin_bit_cast(unsigned, p);
}

__device__ __forceinline__ float sigf(float x) {
    return 1.0f / (1.0f + __expf(-x));
}

__device__ __forceinline__ v8f wmma_f16(v16h a, v16h b, v8f c) {
    return __builtin_amdgcn_wmma_f32_16x16x32_f16(
        false, a, false, b, (short)0, c, false, false);
}

// ---- grid-wide generation barrier (64 resident blocks) ----
__device__ __forceinline__ void grid_barrier(unsigned* bar) {
    __syncthreads();
    if (threadIdx.x == 0) {
        __threadfence();
        unsigned g = __hip_atomic_load(bar + 1, __ATOMIC_ACQUIRE, __HIP_MEMORY_SCOPE_AGENT);
        unsigned a = __hip_atomic_fetch_add(bar, 1u, __ATOMIC_ACQ_REL, __HIP_MEMORY_SCOPE_AGENT);
        if (a == (unsigned)(NBLK - 1)) {
            __hip_atomic_store(bar, 0u, __ATOMIC_RELAXED, __HIP_MEMORY_SCOPE_AGENT);
            __hip_atomic_fetch_add(bar + 1, 1u, __ATOMIC_ACQ_REL, __HIP_MEMORY_SCOPE_AGENT);
        } else {
            while (__hip_atomic_load(bar + 1, __ATOMIC_ACQUIRE, __HIP_MEMORY_SCOPE_AGENT) == g) {
                __builtin_amdgcn_s_sleep(2);
            }
        }
    }
    __syncthreads();
    __threadfence();
}

// ---- pack W (K=512 x N=2048, row-major f32) into WMMA B-operand f16 layout ----
// element((nt,kb,lane,e)) = W[kb*32 + (lane>>4)*16 + e][nt*16 + (lane&15)]
__global__ __launch_bounds__(256) void pack_w(const float* __restrict__ W,
                                              unsigned short* __restrict__ P) {
    int tid  = blockIdx.x * blockDim.x + threadIdx.x;   // 0 .. 65535
    int lane = tid & 31;
    int kb   = (tid >> 5) & 15;
    int nt   = tid >> 9;                                // 0..127 (= g*32 + col_tile)
    int n     = nt * 16 + (lane & 15);
    int kbase = kb * 32 + (lane >> 4) * 16;
    V16U o;
#pragma unroll
    for (int e = 0; e < 16; e += 2)
        o.q[e >> 1] = pk2h(W[(size_t)(kbase + e) * G4H + n],
                           W[(size_t)(kbase + e + 1) * G4H + n]);
    uint4* dst = (uint4*)(P + (size_t)tid * 16);
    dst[0] = o.u[0];
    dst[1] = o.u[1];
}

// ---- init: barrier words, hbuf[0] = mask(resets[0], h0), xbuf[0] = f16(x[0]) ----
__global__ __launch_bounds__(256) void init_state(const float* __restrict__ h0,
                                                  const float* __restrict__ x,
                                                  const unsigned char* __restrict__ resets,
                                                  unsigned short* __restrict__ hbuf,
                                                  unsigned short* __restrict__ xbuf,
                                                  unsigned* __restrict__ bar) {
    int tid = blockIdx.x * blockDim.x + threadIdx.x;    // 16384 threads * 8 elems
    if (tid == 0) { bar[0] = 0u; bar[1] = 0u; }
    int base = tid * 8;
    int row  = base >> 9;
    unsigned char m = resets[row];                      // t = 0
    float4 a = *(const float4*)(h0 + base);
    float4 b = *(const float4*)(h0 + base + 4);
    V16U o;
    o.q[0] = m ? 0u : pk2h(a.x, a.y);
    o.q[1] = m ? 0u : pk2h(a.z, a.w);
    o.q[2] = m ? 0u : pk2h(b.x, b.y);
    o.q[3] = m ? 0u : pk2h(b.z, b.w);
    *(uint4*)(hbuf + base) = o.u[0];
    float4 c = *(const float4*)(x + base);
    float4 d = *(const float4*)(x + base + 4);
    V16U p;
    p.q[0] = pk2h(c.x, c.y);
    p.q[1] = pk2h(c.z, c.w);
    p.q[2] = pk2h(d.x, d.y);
    p.q[3] = pk2h(d.z, d.w);
    *(uint4*)(xbuf + base) = p.u[0];
}

// ---- persistent scan kernel ----
// block b: col_tile = b>>1, row_tile = (b&1)*8 + wave  -> 64 blocks x 8 waves = 512 tiles
// LDS layout: [kb][matrix][gate][lane][16] f16; per-kb window = 8 KB, all DS offsets < 7.3 KB
__global__ __launch_bounds__(NTHR) void lstm_scan(
    const float* __restrict__ x,             // [T][B][H] f32
    const unsigned char* __restrict__ resets,// [T][B]
    const float* __restrict__ c0,            // [B][H]
    const float* __restrict__ bias,          // [4H]
    const unsigned short* __restrict__ WiP,  // packed f16
    const unsigned short* __restrict__ WhP,  // packed f16
    unsigned short* __restrict__ hbuf,       // 2 * [B][H] f16 ping-pong
    unsigned short* __restrict__ xbuf,       // 2 * [B][H] f16 ping-pong
    float* __restrict__ ys,                  // [T][B][H]  (d_out base)
    float* __restrict__ c_f,                 // [B][H]
    float* __restrict__ h_f,                 // [B][H]
    unsigned* __restrict__ bar) {

    extern __shared__ unsigned short smem[];             // 131072 bytes

    const int lane = threadIdx.x & 31;
    const int wave = threadIdx.x >> 5;
    const int col_tile = blockIdx.x >> 1;                // 0..31
    const int row_tile = (blockIdx.x & 1) * 8 + wave;    // 0..15
    const int half = lane >> 4;
    const int l15  = lane & 15;
    const int tid  = blockIdx.x * NTHR + threadIdx.x;    // 0..16383

    // one-time: stage this col_tile's packed weights into LDS, interleaved layout
    {
        uint4* s4 = (uint4*)smem;
        const uint4* gWi = (const uint4*)WiP;
        const uint4* gWh = (const uint4*)WhP;
        for (int d = threadIdx.x; d < 8192; d += NTHR) { // 8192 uint4 = 128 KB
            int j  = d & 63;                             // 64 uint4 per [lane][16] slab
            int g  = (d >> 6) & 3;
            int m  = (d >> 8) & 1;
            int kb = d >> 9;
            const uint4* src = m ? gWh : gWi;
            s4[d] = src[((size_t)(g * 32 + col_tile) * 16 + kb) * 64 + j];
        }
        __syncthreads();
    }

    const int arow  = row_tile * 16 + l15;       // A-operand row
    const int ccol  = col_tile * 16 + l15;       // C/D column
    const int crow0 = row_tile * 16 + half * 8;  // C/D rows crow0..crow0+7

    const float bi  = bias[0 * HH + ccol];
    const float bf_ = bias[1 * HH + ccol];
    const float bg  = bias[2 * HH + ccol];
    const float bo  = bias[3 * HH + ccol];

    float creg[8];
#pragma unroll
    for (int r = 0; r < 8; ++r)
        creg[r] = c0[(size_t)(crow0 + r) * HH + ccol];

    for (int t = 0; t < TT; ++t) {
        const unsigned short* xrow = xbuf + (size_t)(t & 1) * BB * HH + (size_t)arow * HH;
        const unsigned short* hrow = hbuf + (size_t)(t & 1) * BB * HH + (size_t)arow * HH;

        v8f a0 = {}, a1 = {}, a2 = {}, a3 = {};
        const unsigned short* wb = smem + lane * 16;     // advances 4096 ushorts per kb

        // initial A operands (kb = 0)
        int k0 = half * 16;
        V16U Ax, Ah;
        Ax.u[0] = *(const uint4*)(xrow + k0);  Ax.u[1] = *(const uint4*)(xrow + k0 + 8);
        Ah.u[0] = *(const uint4*)(hrow + k0);  Ah.u[1] = *(const uint4*)(hrow + k0 + 8);
        // initial Wi fragments (kb = 0)
        V16U Bi0, Bi1, Bi2, Bi3;
        Bi0.u[0] = *(const uint4*)(wb);         Bi0.u[1] = *(const uint4*)(wb + 8);
        Bi1.u[0] = *(const uint4*)(wb + 512);   Bi1.u[1] = *(const uint4*)(wb + 520);
        Bi2.u[0] = *(const uint4*)(wb + 1024);  Bi2.u[1] = *(const uint4*)(wb + 1032);
        Bi3.u[0] = *(const uint4*)(wb + 1536);  Bi3.u[1] = *(const uint4*)(wb + 1544);

        for (int kb = 0; kb < KB; ++kb) {
            // Wh fragments for this kb (in flight during the Wi WMMAs)
            V16U Bh0, Bh1, Bh2, Bh3;
            Bh0.u[0] = *(const uint4*)(wb + 2048);  Bh0.u[1] = *(const uint4*)(wb + 2056);
            Bh1.u[0] = *(const uint4*)(wb + 2560);  Bh1.u[1] = *(const uint4*)(wb + 2568);
            Bh2.u[0] = *(const uint4*)(wb + 3072);  Bh2.u[1] = *(const uint4*)(wb + 3080);
            Bh3.u[0] = *(const uint4*)(wb + 3584);  Bh3.u[1] = *(const uint4*)(wb + 3592);

            a0 = wmma_f16(Ax.v, Bi0.v, a0);
            a1 = wmma_f16(Ax.v, Bi1.v, a1);
            a2 = wmma_f16(Ax.v, Bi2.v, a2);
            a3 = wmma_f16(Ax.v, Bi3.v, a3);

            // next kb's Wi fragments + A operands (in flight during the Wh WMMAs)
            V16U Bn0, Bn1, Bn2, Bn3, Axn, Ahn;
            if (kb < KB - 1) {
                const unsigned short* wn = wb + 4096;
                Bn0.u[0] = *(const uint4*)(wn);         Bn0.u[1] = *(const uint4*)(wn + 8);
                Bn1.u[0] = *(const uint4*)(wn + 512);   Bn1.u[1] = *(const uint4*)(wn + 520);
                Bn2.u[0] = *(const uint4*)(wn + 1024);  Bn2.u[1] = *(const uint4*)(wn + 1032);
                Bn3.u[0] = *(const uint4*)(wn + 1536);  Bn3.u[1] = *(const uint4*)(wn + 1544);
                const int kn = (kb + 1) * 32 + half * 16;
                Axn.u[0] = *(const uint4*)(xrow + kn);  Axn.u[1] = *(const uint4*)(xrow + kn + 8);
                Ahn.u[0] = *(const uint4*)(hrow + kn);  Ahn.u[1] = *(const uint4*)(hrow + kn + 8);
            }

            a0 = wmma_f16(Ah.v, Bh0.v, a0);
            a1 = wmma_f16(Ah.v, Bh1.v, a1);
            a2 = wmma_f16(Ah.v, Bh2.v, a2);
            a3 = wmma_f16(Ah.v, Bh3.v, a3);

            if (kb < KB - 1) {
                Bi0 = Bn0; Bi1 = Bn1; Bi2 = Bn2; Bi3 = Bn3;
                Ax = Axn; Ah = Ahn;
                wb += 4096;
            }
        }

        // elementwise LSTM update for this wave's 16x16 tile
        const bool last = (t == TT - 1);
        unsigned long long rcur =
            *(const unsigned long long*)(resets + (size_t)t * BB + crow0);
        unsigned long long rnext = 0ull;
        if (!last)
            rnext = *(const unsigned long long*)(resets + (size_t)(t + 1) * BB + crow0);

        float* yt = ys + (size_t)t * BB * HH;
        unsigned short* hnx = hbuf + (size_t)((t + 1) & 1) * BB * HH;
#pragma unroll
        for (int r = 0; r < 8; ++r) {
            float gi = sigf(a0[r] + bi);
            float gf = sigf(a1[r] + bf_);
            float gg = tanhf(a2[r] + bg);
            float go = sigf(a3[r] + bo);
            float cold = ((rcur >> (8 * r)) & 0xFFull) ? 0.0f : creg[r];
            float cn = gf * cold + gi * gg;
            float hn = go * tanhf(cn);
            creg[r] = cn;
            size_t off = (size_t)(crow0 + r) * HH + ccol;
            yt[off] = hn;
            if (!last) {
                hnx[off] = ((rnext >> (8 * r)) & 0xFFull) ? (unsigned short)0 : f2h(hn);
            } else {
                c_f[off] = cn;
                h_f[off] = hn;
            }
        }

        // cooperative f16 conversion of x[t+1] into the other x ping-pong buffer
        if (!last) {
            int base = tid * 8;
            const float* xs = x + (size_t)(t + 1) * BB * HH + base;
            float4 a = *(const float4*)(xs);
            float4 b = *(const float4*)(xs + 4);
            V16U o;
            o.q[0] = pk2h(a.x, a.y);
            o.q[1] = pk2h(a.z, a.w);
            o.q[2] = pk2h(b.x, b.y);
            o.q[3] = pk2h(b.z, b.w);
            *(uint4*)(xbuf + (size_t)((t + 1) & 1) * BB * HH + base) = o.u[0];
        }

        grid_barrier(bar);
    }
}

extern "C" void kernel_launch(void* const* d_in, const int* in_sizes, int n_in,
                              void* d_out, int out_size, void* d_ws, size_t ws_size,
                              hipStream_t stream) {
    const float*         x      = (const float*)d_in[0];
    const unsigned char* resets = (const unsigned char*)d_in[1];
    const float*         c0     = (const float*)d_in[2];
    const float*         h0     = (const float*)d_in[3];
    const float*         Wi     = (const float*)d_in[4];
    const float*         Wh     = (const float*)d_in[5];
    const float*         bias   = (const float*)d_in[6];

    unsigned short* WiP  = (unsigned short*)d_ws;                       // 2 MB
    unsigned short* WhP  = WiP + (size_t)HH * G4H;                      // 2 MB
    unsigned short* hbuf = WhP + (size_t)HH * G4H;                      // 512 KB
    unsigned short* xbuf = hbuf + 2 * (size_t)BB * HH;                  // 512 KB
    unsigned*       bar  = (unsigned*)(xbuf + 2 * (size_t)BB * HH);

    float* ys  = (float*)d_out;
    float* c_f = ys + (size_t)TT * BB * HH;
    float* h_f = c_f + (size_t)BB * HH;

    pack_w<<<256, 256, 0, stream>>>(Wi, WiP);
    pack_w<<<256, 256, 0, stream>>>(Wh, WhP);
    init_state<<<64, 256, 0, stream>>>(h0, x, resets, hbuf, xbuf, bar);
    lstm_scan<<<NBLK, NTHR, LDS_BYTES, stream>>>(x, resets, c0, bias, WiP, WhP,
                                                 hbuf, xbuf, ys, c_f, h_f, bar);
}
